// ScaledDotProductAttention_22428319219843
// MI455X (gfx1250) — compile-verified, matrix operand order checked by reading, not run on hardware
//
#include <hip/hip_runtime.h>

// MI455X / gfx1250, wave32.
// - bf16 WMMA (f32 accumulate) for both QK^T and P@V (bandwidth-bound kernel:
//   the 1.07 GB attn stream dominates; bf16 WMMA pushes compute far under the
//   ~55us HBM floor, fp32 K=4 WMMA would not).
// - Full 16x2048 softmax row-block resident in LDS (CDNA5: 320KB/WGP).
// - attn/out written with non-temporal stores so K/V (128MB total) stay in L2.
// - V staged to LDS via the Tensor Data Mover (tensor_load_to_lds + TENSORcnt),
//   using TDM row padding to de-conflict LDS banks.

typedef __bf16 v16bf __attribute__((ext_vector_type(16)));
typedef float  v8f   __attribute__((ext_vector_type(8)));
typedef unsigned int u32x4 __attribute__((ext_vector_type(4)));
typedef int          i32x4 __attribute__((ext_vector_type(4)));
typedef int          i32x8 __attribute__((ext_vector_type(8)));

#define B_      64
#define S_      2048
#define D_      128
#define QTILE   16
#define NWAVE   8
#define NTHR    256
#define VROWS   32                     // V rows staged per TDM chunk
#define VPAD    4                      // f32 pad per LDS V row (TDM pad_amount)
#define SCALE   0.08838834764831843f   // 1/sqrt(128)

#if __has_builtin(__builtin_amdgcn_tensor_load_to_lds)
#define HAVE_TDM 1
#endif

__launch_bounds__(NTHR, 1)
__global__ void attn_fused_kernel(const float* __restrict__ q,
                                  const float* __restrict__ k,
                                  const float* __restrict__ v,
                                  float* __restrict__ outp,
                                  float* __restrict__ attnp) {
    __shared__ __bf16 sQh[QTILE][D_ + 8];          // Q tile, bf16
    __shared__ float  sL[QTILE][S_ + 4];           // logits -> exp(P) (128KB)
    __shared__ float  sVf[VROWS][D_ + VPAD];       // TDM-staged V chunk (padded)
    __shared__ float  sRed[QTILE][16];
    __shared__ float  sRowMax[QTILE];
    __shared__ float  sRowInv[QTILE];

    const int bb   = blockIdx.x >> 7;              // batch*head index
    const int qt   = blockIdx.x & 127;             // q tile index
    const int tid  = threadIdx.x;
    const int wave = tid >> 5;
    const int lane = tid & 31;
    const int m16  = lane & 15;
    const int hi   = (lane >> 4) & 1;
    const int hi8  = hi * 8;

    const size_t bbase = (size_t)bb * S_ * D_;
    const float* Qg = q + bbase + (size_t)(qt * QTILE) * D_;
    const float* Kg = k + bbase;
    const float* Vg = v + bbase;

    // ---- stage Q tile into LDS as bf16 ----
    for (int i = tid; i < QTILE * D_; i += NTHR) {
        sQh[i >> 7][i & (D_ - 1)] = (__bf16)Qg[i];
    }
    __syncthreads();

    // ---- phase 1: logits L = (Q K^T)*scale, one 16x16 tile per wave-step ----
    for (int kt = wave; kt < S_ / 16; kt += NWAVE) {
        const int n = kt * 16 + m16;               // K-matrix row this lane feeds
        const float* kr = Kg + (size_t)n * D_ + hi8;
        v8f c = {};
        #pragma unroll
        for (int cb = 0; cb < D_; cb += 32) {      // 4 chunks of K=32
            v16bf a, bm;
            const __bf16* q0 = &sQh[m16][cb + hi8];
            const float*  k0 = kr + cb;
            #pragma unroll
            for (int j = 0; j < 8; ++j) {
                a[j]      = q0[j];
                a[j + 8]  = q0[16 + j];
                bm[j]     = (__bf16)k0[j];
                bm[j + 8] = (__bf16)k0[16 + j];
            }
            c = __builtin_amdgcn_wmma_f32_16x16x32_bf16(
                    false, a, false, bm, (short)0, c, false, false);
        }
        const int col = kt * 16 + m16;
        #pragma unroll
        for (int r = 0; r < 8; ++r) {
            sL[r + hi8][col] = c[r] * SCALE;       // C layout: VGPR r -> row r (+8 hi)
        }
    }
    __syncthreads();

    // ---- phase 2: row softmax over 2048 cols (16 threads per row) ----
    const int row = tid >> 4;
    const int sub = tid & 15;
    float mx = -3.4e38f;
    for (int c0 = sub; c0 < S_; c0 += 16) mx = fmaxf(mx, sL[row][c0]);
    sRed[row][sub] = mx;
    __syncthreads();
    if (sub == 0) {
        float m2 = sRed[row][0];
        #pragma unroll
        for (int i = 1; i < 16; ++i) m2 = fmaxf(m2, sRed[row][i]);
        sRowMax[row] = m2;
    }
    __syncthreads();
    {
        const float rmx = sRowMax[row];
        float sum = 0.f;
        for (int c0 = sub; c0 < S_; c0 += 16) {
            float e = __expf(sL[row][c0] - rmx);
            sL[row][c0] = e;
            sum += e;
        }
        sRed[row][sub] = sum;
    }
    __syncthreads();
    if (sub == 0) {
        float s2 = 0.f;
        #pragma unroll
        for (int i = 0; i < 16; ++i) s2 += sRed[row][i];
        sRowInv[row] = 1.0f / s2;
    }
    __syncthreads();

    // ---- phase 3: stream attn = P * inv_sum (non-temporal) ----
    {
        float* at = attnp + (size_t)bb * S_ * S_ + (size_t)(qt * QTILE) * S_;
        for (int i = tid; i < QTILE * S_; i += NTHR) {
            const int r  = i >> 11;
            const int c0 = i & (S_ - 1);
            __builtin_nontemporal_store(sL[r][c0] * sRowInv[r],
                                        at + (size_t)r * S_ + c0);
        }
    }

    // ---- phase 4: out = (P @ V)*inv_sum ; V staged 32 rows at a time ----
    {
        const int dcol = wave * 16 + m16;
        const unsigned sVf_lds = (unsigned)(size_t)(&sVf[0][0]);
        v8f acc = {};
        for (int cb = 0; cb < S_; cb += VROWS) {
            __syncthreads();                       // sVf consumed
#ifdef HAVE_TDM
            if (wave == 0) {
                const unsigned long long ga =
                    (unsigned long long)(size_t)(Vg + (size_t)cb * D_);
                u32x4 g0;
                i32x8 g1;
                i32x4 g2 = {0, 0, 0, 0}, g3 = {0, 0, 0, 0};
                // D# group0: count=1 | lds_addr | global_addr[56:0] | type=2
                g0[0] = 1u;
                g0[1] = sVf_lds;
                g0[2] = (unsigned)(ga & 0xFFFFFFFFu);
                g0[3] = (unsigned)((ga >> 32) & 0x1FFFFFFu) | (2u << 30);
                // D# group1: data_size=4B, pad every 128 DWORDs by 4 DWORDs
                g1[0] = (int)((2u << 16) | (1u << 20) | (6u << 22) | (3u << 25));
                g1[1] = (int)(128u << 16);         // tensor_dim0 = 128
                g1[2] = (int)((unsigned)VROWS << 16); // tensor_dim1 = 32
                g1[3] = (int)(128u << 16);         // tile_dim0 = 128
                g1[4] = VROWS;                     // tile_dim1 = 32
                g1[5] = 128;                       // tensor_dim0_stride = 128
                g1[6] = 0;
                g1[7] = 0;
#if __has_include(<hip/amd_detail/amd_gfx1250_TDM.h>)
                i32x8 g4 = {0, 0, 0, 0, 0, 0, 0, 0};
                __builtin_amdgcn_tensor_load_to_lds(g0, g1, g2, g3, g4, 0);
#else
                __builtin_amdgcn_tensor_load_to_lds(g0, g1, g2, g3, 0);
#endif
                __builtin_amdgcn_s_wait_tensorcnt(0);
            }
#else
            for (int i = tid; i < VROWS * D_; i += NTHR) {
                sVf[i >> 7][i & (D_ - 1)] = Vg[(size_t)(cb + (i >> 7)) * D_ + (i & (D_ - 1))];
            }
#endif
            __syncthreads();                       // sVf published
            v16bf a, bm;
            const float* p0 = &sL[m16][cb + hi8];
            #pragma unroll
            for (int j = 0; j < 8; ++j) {
                a[j]      = (__bf16)p0[j];
                a[j + 8]  = (__bf16)p0[16 + j];
                bm[j]     = (__bf16)sVf[hi8 + j][dcol];
                bm[j + 8] = (__bf16)sVf[hi8 + 16 + j][dcol];
            }
            acc = __builtin_amdgcn_wmma_f32_16x16x32_bf16(
                    false, a, false, bm, (short)0, acc, false, false);
        }
        float* ob = outp + (size_t)bb * S_ * D_ + (size_t)(qt * QTILE) * D_;
        #pragma unroll
        for (int r = 0; r < 8; ++r) {
            const int rr = r + hi8;
            __builtin_nontemporal_store(acc[r] * sRowInv[rr],
                                        ob + (size_t)rr * D_ + dcol);
        }
    }
}

extern "C" void kernel_launch(void* const* d_in, const int* in_sizes, int n_in,
                              void* d_out, int out_size, void* d_ws, size_t ws_size,
                              hipStream_t stream) {
    const float* q = (const float*)d_in[0];
    const float* k = (const float*)d_in[1];
    const float* v = (const float*)d_in[2];
    float* outp  = (float*)d_out;                  // [B,S,D]
    float* attnp = outp + (size_t)B_ * S_ * D_;    // [B,S,S] follows
    (void)in_sizes; (void)n_in; (void)out_size; (void)d_ws; (void)ws_size;

    dim3 grid(B_ * (S_ / QTILE));                  // 8192 workgroups
    dim3 block(NTHR);                              // 8 wave32
    attn_fused_kernel<<<grid, block, 0, stream>>>(q, k, v, outp, attnp);
}
